// Attention_39857296507630
// MI455X (gfx1250) — compile-verified, matrix operand order checked by reading, not run on hardware
//
#include <hip/hip_runtime.h>
#include <hip/hip_bf16.h>

// MI455X (gfx1250) multi-head attention forward, bf16 WMMA path.
// Workspace: ~44 MB (bf16 x, K, Q, Vt, Z, weights + fp32 stats).

typedef __attribute__((ext_vector_type(16))) __bf16 v16bf;
typedef __attribute__((ext_vector_type(8)))  float  v8f;

constexpr int Bn = 2, Sn = 4096, En = 512, An = 8, Hn = 64;
constexpr float SCL = 0.125f;  // 1/sqrt(64)

union frag_u { uint4 q[2]; v16bf v; };

__device__ __forceinline__ unsigned pk2(float a, float b) {
  union { __bf16 h[2]; unsigned u; } r;
  r.h[0] = (__bf16)a; r.h[1] = (__bf16)b; return r.u;
}

// Load a 16x32 bf16 fragment from row-major storage (row stride ld elems).
// ISA 16-bit A layout: lane<16 -> row=lane, K {0..7,16..23};
// lane>=16 -> row=lane-16, K {8..15,24..31}. B fragments use the same loader
// on N-major (i.e. B^T row-major) data.
__device__ __forceinline__ v16bf ldfrag(const __bf16* base, int ld) {
  const int lane = threadIdx.x & 31;
  const int r = lane & 15, hi = lane >> 4;
  const __bf16* p = base + r * ld + hi * 8;
  frag_u u;
  u.q[0] = *(const uint4*)(p);
  u.q[1] = *(const uint4*)(p + 16);
  return u.v;
}

__device__ __forceinline__ v8f wmma_bf16(v16bf a, v16bf b, v8f c) {
  return __builtin_amdgcn_wmma_f32_16x16x32_bf16(false, a, false, b,
                                                 (short)0, c, false, false);
}

// gfx1250 async global->LDS copy (ASYNCcnt-tracked). The LDS byte address for
// VDST is the low 32 bits of the generic pointer (ISA 10.2: LDS aperture
// addresses truncate to the LDS offset).
__device__ __forceinline__ void async_g2l_b128(const __bf16* g, __bf16* l) {
  unsigned ldsoff = (unsigned)(unsigned long long)(uintptr_t)l;
  unsigned long long ga = (unsigned long long)(uintptr_t)g;
  asm volatile("global_load_async_to_lds_b128 %0, %1, off"
               :: "v"(ldsoff), "v"(ga) : "memory");
}
__device__ __forceinline__ void wait_async0() {
  asm volatile("s_wait_asynccnt 0x0" ::: "memory");
}

// ---------------- kernel 0: fp32 -> bf16 bulk convert (8 elems/thread) -----
__global__ void k_cvt8(const float* __restrict__ src, __bf16* __restrict__ dst,
                       int n8) {
  int i = blockIdx.x * blockDim.x + threadIdx.x;
  if (i >= n8) return;
  long base = (long)i * 8;
  float4 v0 = *(const float4*)(src + base);
  float4 v1 = *(const float4*)(src + base + 4);
  uint4 p = { pk2(v0.x, v0.y), pk2(v0.z, v0.w),
              pk2(v1.x, v1.y), pk2(v1.z, v1.w) };
  *(uint4*)(dst + base) = p;
}

// ---------------- kernel 1: K/Q/V projections ------------------------------
// grid (B*A, S/128), 256 thr (8 waves). Wave w: 16 c-rows x 64 h x 3 outputs.
// x fragments come straight from global bf16; the three 64x64 weight chunks
// are staged to LDS with async global->LDS copies each e-step.
__global__ __launch_bounds__(256) void k_proj(
    const __bf16* __restrict__ Xb, const __bf16* __restrict__ WAll,
    __bf16* __restrict__ Kb, __bf16* __restrict__ Qb, __bf16* __restrict__ VT) {
  __shared__ __bf16 wt[3 * 64 * 64];  // 24 KB
  const int bidx = blockIdx.x;        // b*A + a
  const int a = bidx & 7, bb = bidx >> 3;
  const int c0 = blockIdx.y * 128;
  const int tid = threadIdx.x;
  const int w = tid >> 5, lane = tid & 31;
  const int nlo = lane & 15, hi = lane >> 4;

  const v8f vzero = {};
  v8f acc[3][4];
#pragma unroll
  for (int o = 0; o < 3; ++o)
#pragma unroll
    for (int j = 0; j < 4; ++j) acc[o][j] = vzero;

  const long xrow = ((long)bb * Sn + c0 + w * 16) * En;

  for (int e0 = 0; e0 < En; e0 += 64) {
    // async-stage W_{K,Q,V}[a][0..64, e0..e0+64): 1536 x 16B, 6 per thread
#pragma unroll
    for (int t = 0; t < 6; ++t) {
      int idx = t * 256 + tid;
      int o = idx >> 9, rem = idx & 511;
      int row = rem >> 3, q = rem & 7;
      const __bf16* g = WAll + ((long)o * An + a) * Hn * En +
                        (long)row * En + e0 + q * 8;
      async_g2l_b128(g, wt + o * 4096 + row * 64 + q * 8);
    }
    wait_async0();
    __syncthreads();

#pragma unroll
    for (int kc = 0; kc < 64; kc += 32) {
      v16bf af = ldfrag(Xb + xrow + e0 + kc, En);
#pragma unroll
      for (int o = 0; o < 3; ++o) {
        v16bf bfr[4];
#pragma unroll
        for (int j = 0; j < 4; ++j)
          bfr[j] = ldfrag(wt + o * 4096 + (j * 16) * 64 + kc, 64);
#pragma unroll
        for (int j = 0; j < 4; ++j)
          acc[o][j] = wmma_bf16(af, bfr[j], acc[o][j]);
      }
    }
    __syncthreads();  // all waves done reading before next async overwrite
  }

  // epilogue: K,Q row-major [S,H]; V transposed [H,S] (packed b128 stores)
  const long baBase = (long)bidx * Sn * Hn;
  const int crow0 = c0 + w * 16 + hi * 8;
#pragma unroll
  for (int j = 0; j < 4; ++j) {
    int h = j * 16 + nlo;
#pragma unroll
    for (int i = 0; i < 8; ++i)
      Kb[baBase + (long)(crow0 + i) * Hn + h] = (__bf16)acc[0][j][i];
#pragma unroll
    for (int i = 0; i < 8; ++i)
      Qb[baBase + (long)(crow0 + i) * Hn + h] = (__bf16)acc[1][j][i];
    uint4 pv = { pk2(acc[2][j][0], acc[2][j][1]), pk2(acc[2][j][2], acc[2][j][3]),
                 pk2(acc[2][j][4], acc[2][j][5]), pk2(acc[2][j][6], acc[2][j][7]) };
    *(uint4*)(VT + ((long)bidx * Hn + h) * Sn + crow0) = pv;
  }
}

// ---------------- kernel 2: per-row-c softmax stats ------------------------
// s[c,C]=k[c].q[C], mask C<=c, softmax over C. Per-lane online softmax (lane
// only sees C == nlo mod 16), butterfly merge across the 16-lane half at end.
__global__ __launch_bounds__(256) void k_stats(
    const __bf16* __restrict__ Kb, const __bf16* __restrict__ Qb,
    float* __restrict__ Mst, float* __restrict__ Rinv) {
  const int bidx = blockIdx.x;
  const int tid = threadIdx.x;
  const int w = tid >> 5, lane = tid & 31;
  const int nlo = lane & 15, hi = lane >> 4;
  const int crow = blockIdx.y * 128 + w * 16;
  const long base = (long)bidx * Sn * Hn;

  v16bf ka0 = ldfrag(Kb + base + (long)crow * Hn, Hn);
  v16bf ka1 = ldfrag(Kb + base + (long)crow * Hn + 32, Hn);

  float m8[8], l8[8];
#pragma unroll
  for (int i = 0; i < 8; ++i) { m8[i] = -1e30f; l8[i] = 0.f; }

  const v8f vzero = {};
  const int cmax = crow + 15;
  for (int C0 = 0; C0 <= cmax; C0 += 16) {
    v16bf qb0 = ldfrag(Qb + base + (long)C0 * Hn, Hn);
    v16bf qb1 = ldfrag(Qb + base + (long)C0 * Hn + 32, Hn);
    v8f s = wmma_bf16(ka0, qb0, vzero);
    s = wmma_bf16(ka1, qb1, s);
    const int Ccol = C0 + nlo;
#pragma unroll
    for (int i = 0; i < 8; ++i) {
      int c = crow + hi * 8 + i;
      float sv = (Ccol > c) ? -3e38f : s[i] * SCL;  // masked -> exp == 0
      float nm = fmaxf(m8[i], sv);
      l8[i] = l8[i] * __expf(m8[i] - nm) + __expf(sv - nm);
      m8[i] = nm;
    }
  }
#pragma unroll
  for (int i = 0; i < 8; ++i) {  // merge the 16 lanes of this half
    float m = m8[i], l = l8[i];
#pragma unroll
    for (int d = 1; d < 16; d <<= 1) {
      float om = __shfl_xor(m, d, 32);
      float ol = __shfl_xor(l, d, 32);
      float nm = fmaxf(m, om);
      l = l * __expf(m - nm) + ol * __expf(om - nm);
      m = nm;
    }
    m8[i] = m; l8[i] = l;
  }
  if (nlo == 0) {
    long sb = (long)bidx * Sn + crow + hi * 8;
#pragma unroll
    for (int i = 0; i < 8; ++i) {
      Mst[sb + i] = m8[i];
      Rinv[sb + i] = 1.0f / l8[i];
    }
  }
}

// ---------------- kernel 3: z[C,h] = sum_{c>=C} p[c,C] v[c,h] --------------
// Recompute s-tiles, p = exp(s/8 - m_c)/l_c, lane-local repack of the s-tile
// accumulator into the p^T A-fragment, then z += p^T x V^T via WMMA.
__global__ __launch_bounds__(256) void k_zpass(
    const __bf16* __restrict__ Kb, const __bf16* __restrict__ Qb,
    const __bf16* __restrict__ VT, const float* __restrict__ Mst,
    const float* __restrict__ Rinv, __bf16* __restrict__ Zb) {
  const int bidx = blockIdx.x;
  const int tid = threadIdx.x;
  const int w = tid >> 5, lane = tid & 31;
  const int nlo = lane & 15, hi = lane >> 4;
  const int C0 = blockIdx.y * 128 + w * 16;
  const long base = (long)bidx * Sn * Hn;
  const long vbase = (long)bidx * Hn * Sn;
  const long sbase = (long)bidx * Sn;

  v16bf qf0 = ldfrag(Qb + base + (long)C0 * Hn, Hn);
  v16bf qf1 = ldfrag(Qb + base + (long)C0 * Hn + 32, Hn);
  const v8f vzero = {};
  v8f z[4];
#pragma unroll
  for (int j = 0; j < 4; ++j) z[j] = vzero;

  const int Ccol = C0 + nlo;
  for (int cb = C0 & ~31; cb < Sn; cb += 32) {
    v16bf pfrag;
#pragma unroll
    for (int t = 0; t < 2; ++t) {
      const int cr = cb + t * 16;
      v16bf kf0 = ldfrag(Kb + base + (long)cr * Hn, Hn);
      v16bf kf1 = ldfrag(Kb + base + (long)cr * Hn + 32, Hn);
      v8f s = wmma_bf16(kf0, qf0, vzero);
      s = wmma_bf16(kf1, qf1, s);
#pragma unroll
      for (int i = 0; i < 8; ++i) {
        int c = cr + hi * 8 + i;
        float p = 0.f;
        if (c >= Ccol) {
          float mc = Mst[sbase + c];
          float rc = Rinv[sbase + c];
          p = __expf(s[i] * SCL - mc) * rc;
        }
        pfrag[t * 8 + i] = (__bf16)p;  // lane-local acc -> A-frag repack
      }
    }
#pragma unroll
    for (int j = 0; j < 4; ++j) {
      v16bf vf = ldfrag(VT + vbase + (long)(j * 16) * Sn + cb, Sn);
      z[j] = wmma_bf16(pfrag, vf, z[j]);
    }
    __builtin_prefetch(Kb + base + (long)(cb + 32) * Hn, 0, 1);
  }
#pragma unroll
  for (int j = 0; j < 4; ++j) {
    int h = j * 16 + nlo;
#pragma unroll
    for (int i = 0; i < 8; ++i)
      Zb[base + (long)(C0 + hi * 8 + i) * Hn + h] = (__bf16)z[j][i];
  }
}

// ---------------- kernel 4: out[b,c,e] = sum_{a,h} W_O[a,e,h] z[b,a,c,h] ---
__global__ __launch_bounds__(256) void k_outproj(
    const __bf16* __restrict__ Zb, const __bf16* __restrict__ WOb,
    float* __restrict__ out) {
  const int b = blockIdx.x;
  const int tid = threadIdx.x;
  const int w = tid >> 5, lane = tid & 31;
  const int nlo = lane & 15, hi = lane >> 4;
  const int c0 = blockIdx.y * 128 + w * 16;
  const int e0 = blockIdx.z * 64;

  const v8f vzero = {};
  v8f acc[4];
#pragma unroll
  for (int j = 0; j < 4; ++j) acc[j] = vzero;

  for (int kk = 0; kk < 16; ++kk) {  // K = A*H = 512, 32-chunks within a head
    const int a = kk >> 1;
    const int h0 = (kk & 1) * 32;
    v16bf af = ldfrag(Zb + (((long)b * An + a) * Sn + c0) * Hn + h0, Hn);
    v16bf bfr[4];
#pragma unroll
    for (int j = 0; j < 4; ++j)
      bfr[j] = ldfrag(WOb + ((long)a * En + e0 + j * 16) * Hn + h0, Hn);
#pragma unroll
    for (int j = 0; j < 4; ++j) acc[j] = wmma_bf16(af, bfr[j], acc[j]);
  }
#pragma unroll
  for (int j = 0; j < 4; ++j) {
    int e = e0 + j * 16 + nlo;
#pragma unroll
    for (int i = 0; i < 8; ++i)
      out[((long)b * Sn + c0 + hi * 8 + i) * En + e] = acc[j][i];
  }
}

extern "C" void kernel_launch(void* const* d_in, const int* in_sizes, int n_in,
                              void* d_out, int out_size, void* d_ws, size_t ws_size,
                              hipStream_t stream) {
  const float* x  = (const float*)d_in[0];
  const float* wk = (const float*)d_in[1];
  const float* wq = (const float*)d_in[2];
  const float* wv = (const float*)d_in[3];
  const float* wo = (const float*)d_in[4];
  float* out = (float*)d_out;

  char* ws = (char*)d_ws;
  size_t off = 0;
  auto carve = [&](size_t bytes) -> char* {
    char* p = ws + off;
    off = (off + bytes + 255) & ~(size_t)255;
    return p;
  };
  const size_t nBA = (size_t)Bn * An;
  const size_t nW  = (size_t)An * Hn * En;          // one projection weight
  __bf16* Xb   = (__bf16*)carve((size_t)Bn * Sn * En * 2);
  __bf16* WAll = (__bf16*)carve(3 * nW * 2);        // K,Q,V weights contiguous
  __bf16* WOb  = (__bf16*)carve(nW * 2);
  __bf16* Kb   = (__bf16*)carve(nBA * Sn * Hn * 2);
  __bf16* Qb   = (__bf16*)carve(nBA * Sn * Hn * 2);
  __bf16* VT   = (__bf16*)carve(nBA * Hn * Sn * 2);
  __bf16* Zb   = (__bf16*)carve(nBA * Sn * Hn * 2);
  float*  Mst  = (float*)carve(nBA * Sn * 4);
  float*  Riv  = (float*)carve(nBA * Sn * 4);
  (void)ws_size; (void)in_sizes; (void)n_in; (void)out_size;

  const int nx8 = Bn * Sn * En / 8, nw8 = (int)(nW / 8);
  k_cvt8<<<dim3((nx8 + 255) / 256), dim3(256), 0, stream>>>(x, Xb, nx8);
  k_cvt8<<<dim3((nw8 + 255) / 256), dim3(256), 0, stream>>>(wk, WAll, nw8);
  k_cvt8<<<dim3((nw8 + 255) / 256), dim3(256), 0, stream>>>(wq, WAll + nW, nw8);
  k_cvt8<<<dim3((nw8 + 255) / 256), dim3(256), 0, stream>>>(wv, WAll + 2 * nW, nw8);
  k_cvt8<<<dim3((nw8 + 255) / 256), dim3(256), 0, stream>>>(wo, WOb, nw8);

  k_proj<<<dim3(Bn * An, Sn / 128), dim3(256), 0, stream>>>(Xb, WAll, Kb, Qb, VT);
  k_stats<<<dim3(Bn * An, Sn / 128), dim3(256), 0, stream>>>(Kb, Qb, Mst, Riv);
  k_zpass<<<dim3(Bn * An, Sn / 128), dim3(256), 0, stream>>>(
      Kb, Qb, VT, Mst, Riv, Zb);
  k_outproj<<<dim3(Bn, Sn / 128, En / 64), dim3(256), 0, stream>>>(Zb, WOb, out);
}
